// MultiHeadAttention_43946105373007
// MI455X (gfx1250) — compile-verified, matrix operand order checked by reading, not run on hardware
//
#include <hip/hip_runtime.h>
#include <hip/hip_bf16.h>

// ---------------------------------------------------------------------------
// MHA forward for MI455X (gfx1250, wave32, WMMA, async Global->LDS).
//   B=4, S=2048, D_MODEL=512, H=8, DK=DV=64, SCALE=0.125
// Outputs: out (B,S,512) f32 then attn (B,H,S,S) f32, concatenated in d_out.
// Memory-bound on the 512MB attn store -> fuse scores+softmax+PV per q-tile.
// ---------------------------------------------------------------------------

typedef __attribute__((ext_vector_type(16))) _Float16 v16h;
typedef __attribute__((ext_vector_type(8)))  float    v8f;
typedef int v4i __attribute__((vector_size(16)));

#define S_LEN 2048
#define DMODEL 512
#define NHEADS 8
#define DK 64
#define SCALE 0.125f

#ifndef __has_builtin
#define __has_builtin(x) 0
#endif

#if defined(__HIP_DEVICE_COMPILE__) && \
    __has_builtin(__builtin_amdgcn_global_load_async_to_lds_b128)
#define ASYNC_LDS 1
#else
#define ASYNC_LDS 0
#endif

// Copy 8 consecutive f32 global -> LDS. On CDNA5 use the async-DMA path
// (GLOBAL_LOAD_ASYNC_TO_LDS_B128, ASYNCcnt); note INST_OFFSET is added to
// BOTH the LDS and global addresses, so two ops with offsets 0/16 move 32B.
__device__ __forceinline__ void async_copy8(float* lds, const float* g) {
#if ASYNC_LDS
  auto gp = (__attribute__((address_space(1))) v4i*)g;   // global src
  auto lp = (__attribute__((address_space(3))) v4i*)lds; // LDS dst
  __builtin_amdgcn_global_load_async_to_lds_b128(gp, lp, 0, 0);
  __builtin_amdgcn_global_load_async_to_lds_b128(gp, lp, 16, 0);
#else
  ((float4*)lds)[0] = ((const float4*)g)[0];
  ((float4*)lds)[1] = ((const float4*)g)[1];
#endif
}

__device__ __forceinline__ void async_join() {
#if ASYNC_LDS
#if __has_builtin(__builtin_amdgcn_s_wait_asynccnt)
  __builtin_amdgcn_s_wait_asynccnt(0);
#else
  asm volatile("s_wait_asynccnt 0" ::: "memory");
#endif
#endif
}

// Build a WMMA 16x16x32 f16 operand fragment from fp32 memory (global or LDS).
// Per ISA 7.12.2 (16-bit A 16x32): lane L holds row/col (L&15); halves 0..7
// are K = k0..k0+7 and halves 8..15 are K = k0+16..k0+23, k0 = (L>=16)*8.
// p0 -> element K=k0, p1 -> element K=k0+16 (both contiguous runs of 8 f32).
__device__ __forceinline__ v16h make_frag(const float* p0, const float* p1) {
  v16h f;
#pragma unroll
  for (int i = 0; i < 8; ++i) f[i] = (_Float16)p0[i];
#pragma unroll
  for (int i = 0; i < 8; ++i) f[8 + i] = (_Float16)p1[i];
  return f;
}

// ---------------------------------------------------------------------------
// Y[M][N] = X[M][K] * W[N][K]^T   (einsum 'bsd,ed->bse')
// Block = 256 threads (8 waves) computing a 64x64 tile. Per 32-wide K chunk,
// A(64x32) and B(64x32) are async-DMA'd into LDS; each wave runs 2 WMMAs
// (one 16x16 m-subtile x two n-subtiles), giving 8x LDS reuse of both tiles.
// ---------------------------------------------------------------------------
__global__ __launch_bounds__(256) void gemm_xwt(const float* __restrict__ X,
                                                const float* __restrict__ W,
                                                float* __restrict__ Y,
                                                int M, int N, int K) {
  __shared__ float As[64][32];
  __shared__ float Bs[64][32];

  const int tilesN = N >> 6;
  const int tM = blockIdx.x / tilesN;
  const int tN = blockIdx.x - tM * tilesN;
  const int row0 = tM << 6, col0 = tN << 6;
  if (row0 >= M) return;

  const int wave = threadIdx.x >> 5;
  const int lane = threadIdx.x & 31;
  const int r    = lane & 15;          // row (A) / col (B) within subtile
  const int k0h  = (lane >> 4) << 3;   // 0 or 8: K sub-offset per lane half
  const int mhi  = k0h;                // C/D: VGPR i holds row i + mhi
  const int mi   = (wave & 3) << 4;    // m subtile base (0,16,32,48)
  const int ni   = (wave >> 2) << 5;   // n subtile pair base (0,32)

  const int ldRow = threadIdx.x >> 2;        // 0..63
  const int ldCol = (threadIdx.x & 3) << 3;  // 0,8,16,24

  const float* xrow = X + (size_t)(row0 + ldRow) * K + ldCol;
  const float* wrow = W + (size_t)(col0 + ldRow) * K + ldCol;

  v8f acc0 = {}, acc1 = {};
  for (int k = 0; k < K; k += 32) {
    async_copy8(&As[ldRow][ldCol], xrow + k);
    async_copy8(&Bs[ldRow][ldCol], wrow + k);
    __builtin_prefetch(xrow + k + 64, 0, 1);   // global_prefetch_b8
    async_join();
    __syncthreads();

    v16h a  = make_frag(&As[mi + r][k0h],      &As[mi + r][k0h + 16]);
    v16h b0 = make_frag(&Bs[ni + r][k0h],      &Bs[ni + r][k0h + 16]);
    v16h b1 = make_frag(&Bs[ni + 16 + r][k0h], &Bs[ni + 16 + r][k0h + 16]);
    acc0 = __builtin_amdgcn_wmma_f32_16x16x32_f16(false, a, false, b0,
                                                  (short)0, acc0, false, false);
    acc1 = __builtin_amdgcn_wmma_f32_16x16x32_f16(false, a, false, b1,
                                                  (short)0, acc1, false, false);
    __syncthreads();
  }

#pragma unroll
  for (int i = 0; i < 8; ++i) {
    const size_t row = (size_t)(row0 + mi + mhi + i) * N + col0 + ni + r;
    Y[row]      = acc0[i];
    Y[row + 16] = acc1[i];
  }
}

// ---------------------------------------------------------------------------
// Vt[b][h][d][s] = Vp[b*S+s][h*64+d]  -> makes PV B-fragments contiguous.
// ---------------------------------------------------------------------------
__global__ __launch_bounds__(256) void transpose_v(const float* __restrict__ Vp,
                                                   float* __restrict__ Vt) {
  const size_t idx = (size_t)blockIdx.x * 256 + threadIdx.x;  // B*H*DK*S
  const int s = idx & (S_LEN - 1);
  const int d = (idx >> 11) & 63;
  const int h = (idx >> 17) & 7;
  const int b = (int)(idx >> 20);
  Vt[idx] = Vp[((size_t)b * S_LEN + s) * DMODEL + h * DK + d];
}

// ---------------------------------------------------------------------------
// Fused attention for one (b, h, q-tile of 16 rows).
// Workgroup = 256 threads (8 waves); wave w owns key columns [w*256, w*256+256).
// Score tile 16x2048 f32 lives in LDS (128KB; CDNA5 WGP has 320KB).
// ---------------------------------------------------------------------------
__global__ __launch_bounds__(256) void attention(
    const float* __restrict__ Qp,   // (B*S, 512), head h at col h*64
    const float* __restrict__ Kp,
    const float* __restrict__ Vt,   // (B,H,64,S) transposed values
    const unsigned char* __restrict__ mask,  // (B,S,S) bool, true => masked
    float* __restrict__ attn,       // (B,H,S,S)
    float* __restrict__ res) {      // (B*S, 512) concat-head output
  __shared__ float sc[16][S_LEN];       // 128 KB score/prob tile
  __shared__ float red[16][17];         // softmax reductions
  __shared__ float outacc[16][DK];      // cross-wave PV accumulator

  const int bi   = blockIdx.x;
  const int qt   = bi & 127;
  const int h    = (bi >> 7) & 7;
  const int b    = bi >> 10;
  const int row0 = qt << 4;
  const size_t bS = (size_t)b * S_LEN;

  for (int i = threadIdx.x; i < 16 * DK; i += 256)
    ((float*)outacc)[i] = 0.0f;

  const int wave  = threadIdx.x >> 5;
  const int lane  = threadIdx.x & 31;
  const int r     = lane & 15;
  const int k0h   = (lane >> 4) << 3;
  const int mhi   = k0h;
  const int cbase = wave << 8;          // 256 key columns per wave

  // ---- Phase 1: scores = clip(Q K^T * scale), mask, into LDS --------------
  const float* qrow = Qp + (bS + row0 + r) * DMODEL + h * DK;
  v16h aq0 = make_frag(qrow + k0h,      qrow + k0h + 16);       // d = 0..31
  v16h aq1 = make_frag(qrow + 32 + k0h, qrow + 32 + k0h + 16);  // d = 32..63

  for (int ct = 0; ct < 16; ++ct) {
    const int col0 = cbase + (ct << 4);
    const float* krow = Kp + (bS + col0 + r) * DMODEL + h * DK;
    v16h bk0 = make_frag(krow + k0h,      krow + k0h + 16);
    v16h bk1 = make_frag(krow + 32 + k0h, krow + 32 + k0h + 16);
    v8f acc = {};
    acc = __builtin_amdgcn_wmma_f32_16x16x32_f16(false, aq0, false, bk0,
                                                 (short)0, acc, false, false);
    acc = __builtin_amdgcn_wmma_f32_16x16x32_f16(false, aq1, false, bk1,
                                                 (short)0, acc, false, false);
#pragma unroll
    for (int i = 0; i < 8; ++i) {
      const int m = mhi + i;
      float sv = fminf(fmaxf(acc[i] * SCALE, -1.0e9f), 1.0e10f);
      if (mask[((size_t)b * S_LEN + row0 + m) * S_LEN + col0 + r]) sv = -1.0e9f;
      sc[m][col0 + r] = sv;
    }
  }
  __syncthreads();

  // ---- Phase 2: row softmax over 2048, clip, write attn to global ---------
  const int row = threadIdx.x >> 4;   // 16 threads per row
  const int sub = threadIdx.x & 15;   // each covers 128 columns
  const int c0  = sub << 7;

  float lm = -1.0e30f;
  for (int c = c0; c < c0 + 128; ++c) lm = fmaxf(lm, sc[row][c]);
  red[row][sub] = lm;
  __syncthreads();
  if (sub == 0) {
    float mm = red[row][0];
#pragma unroll
    for (int i = 1; i < 16; ++i) mm = fmaxf(mm, red[row][i]);
    red[row][16] = mm;
  }
  __syncthreads();
  const float rmax = red[row][16];
  float ls = 0.0f;
  for (int c = c0; c < c0 + 128; ++c) {
    float e = __expf(sc[row][c] - rmax);
    sc[row][c] = e;
    ls += e;
  }
  __syncthreads();
  red[row][sub] = ls;
  __syncthreads();
  if (sub == 0) {
    float ss = red[row][0];
#pragma unroll
    for (int i = 1; i < 16; ++i) ss += red[row][i];
    red[row][16] = ss;
  }
  __syncthreads();
  const float inv = 1.0f / red[row][16];
  float* arow = attn + (((size_t)(b * NHEADS + h) * S_LEN) + row0 + row) * S_LEN;
  for (int c = c0; c < c0 + 128; ++c) {
    float p = fminf(fmaxf(sc[row][c] * inv, 1.0e-8f), 1.0f);
    sc[row][c] = p;
    arow[c] = p;
  }
  __syncthreads();

  // ---- Phase 3: head_out[16][64] = P @ V, WMMA over LDS probs -------------
  // B-fragment: lane's column d = nt*16 + r; K (= key pos) runs contiguously
  // in Vt, so both 8-element half-runs are contiguous 32B loads.
  const float* vhead = Vt + (size_t)(b * NHEADS + h) * DK * S_LEN;
  v8f acc4[4] = {};
  for (int kc = 0; kc < 256; kc += 32) {
    const float* pa = &sc[r][cbase + kc + k0h];
    v16h ap = make_frag(pa, pa + 16);
#pragma unroll
    for (int nt = 0; nt < 4; ++nt) {
      const float* pv = vhead + (size_t)((nt << 4) + r) * S_LEN
                        + cbase + kc + k0h;
      v16h bv = make_frag(pv, pv + 16);
      acc4[nt] = __builtin_amdgcn_wmma_f32_16x16x32_f16(
          false, ap, false, bv, (short)0, acc4[nt], false, false);
    }
  }
#pragma unroll
  for (int nt = 0; nt < 4; ++nt)
#pragma unroll
    for (int i = 0; i < 8; ++i)
      atomicAdd(&outacc[mhi + i][(nt << 4) + r], acc4[nt][i]);  // ds_add_f32
  __syncthreads();

  for (int idx = threadIdx.x; idx < 16 * DK; idx += 256) {
    const int rr = idx >> 6, dd = idx & 63;
    res[(bS + row0 + rr) * DMODEL + h * DK + dd] = outacc[rr][dd];
  }
}

// ---------------------------------------------------------------------------
// out = layernorm(fc + resid) * gamma + beta ; one block per row of 512.
// ---------------------------------------------------------------------------
__global__ __launch_bounds__(256) void ln_residual(
    const float* __restrict__ fc, const float* __restrict__ resid,
    const float* __restrict__ gamma, const float* __restrict__ beta,
    float* __restrict__ out) {
  __shared__ float redbuf[256];
  const size_t row = blockIdx.x;
  const int t = threadIdx.x;
  const float* xr = fc + row * DMODEL;
  const float* rr = resid + row * DMODEL;

  float v0 = xr[t] + rr[t];
  float v1 = xr[t + 256] + rr[t + 256];

  redbuf[t] = v0 + v1;
  __syncthreads();
  for (int s = 128; s > 0; s >>= 1) {
    if (t < s) redbuf[t] += redbuf[t + s];
    __syncthreads();
  }
  const float mu = redbuf[0] * (1.0f / DMODEL);
  __syncthreads();

  const float d0 = v0 - mu, d1 = v1 - mu;
  redbuf[t] = d0 * d0 + d1 * d1;
  __syncthreads();
  for (int s = 128; s > 0; s >>= 1) {
    if (t < s) redbuf[t] += redbuf[t + s];
    __syncthreads();
  }
  const float inv = rsqrtf(redbuf[0] * (1.0f / DMODEL) + 1.0e-5f);

  out[row * DMODEL + t]       = d0 * inv * gamma[t] + beta[t];
  out[row * DMODEL + t + 256] = d1 * inv * gamma[t + 256] + beta[t + 256];
}

// ---------------------------------------------------------------------------
extern "C" void kernel_launch(void* const* d_in, const int* in_sizes, int n_in,
                              void* d_out, int out_size, void* d_ws,
                              size_t ws_size, hipStream_t stream) {
  (void)in_sizes; (void)n_in; (void)out_size; (void)ws_size;

  const float* inQ = (const float*)d_in[0];
  const float* inK = (const float*)d_in[1];
  const float* inV = (const float*)d_in[2];
  const unsigned char* mask = (const unsigned char*)d_in[3];  // bool array
  const float* WQ  = (const float*)d_in[4];
  const float* WK  = (const float*)d_in[5];
  const float* WV  = (const float*)d_in[6];
  const float* Wfc = (const float*)d_in[7];
  const float* g   = (const float*)d_in[8];
  const float* be  = (const float*)d_in[9];

  const int B = 4, M = B * S_LEN;               // 8192 rows
  const size_t plane = (size_t)M * DMODEL;      // 4,194,304 floats

  float* ws  = (float*)d_ws;
  float* Qp  = ws;
  float* Kp  = ws + plane;
  float* Vp  = ws + 2 * plane;
  float* res = ws + 3 * plane;
  float* fco = ws + 4 * plane;
  float* Vt  = ws + 5 * plane;                  // (B,H,64,S)

  float* out  = (float*)d_out;
  float* attn = out + plane;                    // (B,H,S,S) region

  // Projections: 64x64 block tiles -> (8192/64)*(512/64) = 1024 blocks.
  const int tiles = (M / 64) * (DMODEL / 64);
  dim3 gB(256), gG(tiles);
  gemm_xwt<<<gG, gB, 0, stream>>>(inQ, WQ, Qp, M, DMODEL, DMODEL);
  gemm_xwt<<<gG, gB, 0, stream>>>(inK, WK, Kp, M, DMODEL, DMODEL);
  gemm_xwt<<<gG, gB, 0, stream>>>(inV, WV, Vp, M, DMODEL, DMODEL);

  transpose_v<<<dim3((unsigned)(plane / 256)), dim3(256), 0, stream>>>(Vp, Vt);

  // Fused attention: B*H*(S/16) = 4096 workgroups of 256.
  attention<<<dim3(4096), dim3(256), 0, stream>>>(Qp, Kp, Vt, mask, attn, res);

  // Output projection + LN(residual).
  gemm_xwt<<<gG, gB, 0, stream>>>(res, Wfc, fco, M, DMODEL, DMODEL);
  ln_residual<<<dim3(M), dim3(256), 0, stream>>>(fco, inQ, g, be, out);
}
